// DirectionalConv3d_28982439313416
// MI455X (gfx1250) — compile-verified, matrix operand order checked by reading, not run on hardware
//
#include <hip/hip_runtime.h>

typedef __attribute__((ext_vector_type(16))) __bf16 v16bf;
typedef __attribute__((ext_vector_type(8)))  __bf16 v8bf;
typedef __attribute__((ext_vector_type(8)))  float  v8f;

#define B_    8
#define CIN   64
#define COUT  64
#define T_    32
#define R_    32
#define CO_   32
#define S_    (T_ * R_ * CO_)   /* 32768 spatial positions per (b, channel) */
#define CS_   (CIN * S_)        /* per-batch x stride in floats */

// Each block: 4 waves (128 threads). Each wave: one full CO row (32 positions
// = 2 WMMA N-tiles), all 64 output channels (4 M-tiles). Weights (7 x 64x64)
// staged once to LDS as bf16; each A-fragment (2x ds_load_b128, exact CDNA5
// 16-bit A layout) feeds 2 WMMAs (both N-tiles). B-fragments gathered per-lane
// from global f32 (hits L2: whole x fits in 192MB L2) and packed to bf16 with
// v_cvt_pk_bf16_f32. Tap validity is forced scalar via readfirstlane so EXEC
// stays all-1s across every v_wmma (ISA requirement).
__global__ __launch_bounds__(128)
void dconv3d_wmma_bf16(const float* __restrict__ x,
                       const float* __restrict__ w0, const float* __restrict__ w1,
                       const float* __restrict__ w2, const float* __restrict__ w3,
                       const float* __restrict__ w4, const float* __restrict__ w5,
                       const float* __restrict__ w6,
                       float* __restrict__ out)
{
    __shared__ __align__(16) __bf16 wlds[7 * COUT * CIN];   // 56 KB

    // ---- stage all 7 weight matrices into LDS as bf16 ----
    {
        const float* wsrc[7] = {w0, w1, w2, w3, w4, w5, w6};
        for (int idx = threadIdx.x; idx < 7 * COUT * CIN; idx += 128) {
            const int tap = idx >> 12;        // COUT*CIN == 4096
            const int e   = idx & 4095;
            wlds[idx] = (__bf16)wsrc[tap][e];
        }
    }
    __syncthreads();

    const int lane = threadIdx.x & 31;
    const int wave = threadIdx.x >> 5;
    const int col  = lane & 15;               // N column of this lane
    const int hi8  = (lane >> 4) << 3;        // 0 (lanes 0-15) or 8 (lanes 16-31)

    const int b  = blockIdx.x >> 8;                          // 256 blocks/batch
    const int s0 = ((blockIdx.x & 255) << 7) + (wave << 5);  // row start (32 pos)
    const int s  = s0 + col;                                 // N-tile 0 position

    // wave-uniform t/r of this row, forced into SGPRs -> scalar branches only
    const int t0u = __builtin_amdgcn_readfirstlane(s0 >> 10);
    const int r0u = __builtin_amdgcn_readfirstlane((s0 >> 5) & (R_ - 1));

    const float* xb = x + (size_t)b * CS_;

    v8f acc[8];   // [mt][ntile]: acc[2*mt+0] = cols 0-15, acc[2*mt+1] = cols 16-31
    #pragma unroll
    for (int i = 0; i < 8; ++i) acc[i] = {};

    const int offs[7] = {0, -(R_ * CO_), (R_ * CO_), -CO_, CO_, -1, 1};

    #pragma unroll
    for (int tap = 0; tap < 7; ++tap) {
        // scalar (SGPR) tile validity for t/r taps -> s_cbranch, EXEC untouched
        if (tap == 1 && t0u == 0)       continue;
        if (tap == 2 && t0u == T_ - 1)  continue;
        if (tap == 3 && r0u == 0)       continue;
        if (tap == 4 && r0u == R_ - 1)  continue;

        // per-lane validity only for the c +/- 1 taps (row = cols 0..31):
        //  tap 5 (c-1): only col 0 of N-tile 0 is out of range
        //  tap 6 (c+1): only col 15 of N-tile 1 is out of range
        bool v0 = true, v1 = true;
        if (tap == 5) v0 = (col > 0);
        if (tap == 6) v1 = (col < 15);

        const int sA = v0 ? (s + offs[tap])      : s;         // clamped in-bounds
        const int sB = v1 ? (s + 16 + offs[tap]) : (s + 16);

        #pragma unroll
        for (int kc = 0; kc < 2; ++kc) {
            const int kb = kc * 32 + hi8;     // channel base for this half-wave
            const float* pA = xb + (size_t)kb * S_ + (size_t)sA;
            const float* pB = xb + (size_t)kb * S_ + (size_t)sB;

            // ---- two B fragments: 16 channel values at each lane's column ----
            float f0[16], f1[16];
            #pragma unroll
            for (int j = 0; j < 8; ++j) {
                f0[j]     = pA[(size_t)j * S_];          // K = kb..kb+7
                f0[8 + j] = pA[(size_t)(16 + j) * S_];   // K = kb+16..kb+23
                f1[j]     = pB[(size_t)j * S_];
                f1[8 + j] = pB[(size_t)(16 + j) * S_];
            }
            v16bf Bm0, Bm1;
            #pragma unroll
            for (int j = 0; j < 16; ++j) {
                const float a = v0 ? f0[j] : 0.0f;
                const float c = v1 ? f1[j] : 0.0f;
                Bm0[j] = (__bf16)a;
                Bm1[j] = (__bf16)c;
            }

            // ---- each A fragment feeds both N-tiles (2 WMMAs / ds pair) ----
            #pragma unroll
            for (int mt = 0; mt < 4; ++mt) {
                const int m = mt * 16 + col;
                const __bf16* wp = &wlds[((tap * COUT) + m) * CIN + kb];
                v8bf a0 = *(const v8bf*)(wp);        // W[m, kb..kb+7]
                v8bf a1 = *(const v8bf*)(wp + 16);   // W[m, kb+16..kb+23]
                v16bf Am = __builtin_shufflevector(
                    a0, a1, 0, 1, 2, 3, 4, 5, 6, 7,
                    8, 9, 10, 11, 12, 13, 14, 15);
                acc[2 * mt + 0] = __builtin_amdgcn_wmma_f32_16x16x32_bf16(
                    false, Am, false, Bm0, (short)0, acc[2 * mt + 0], false, false);
                acc[2 * mt + 1] = __builtin_amdgcn_wmma_f32_16x16x32_bf16(
                    false, Am, false, Bm1, (short)0, acc[2 * mt + 1], false, false);
            }
        }
    }

    // ---- store: lane holds N=col / col+16, M = mt*16 + hi8 + v ----
    float* ob = out + (size_t)b * (size_t)(COUT * S_) + (size_t)s0;
    #pragma unroll
    for (int mt = 0; mt < 4; ++mt) {
        #pragma unroll
        for (int v = 0; v < 8; ++v) {
            const size_t mrow = (size_t)(mt * 16 + hi8 + v) * S_;
            ob[mrow + col]      = acc[2 * mt + 0][v];
            ob[mrow + col + 16] = acc[2 * mt + 1][v];
        }
    }
}

extern "C" void kernel_launch(void* const* d_in, const int* in_sizes, int n_in,
                              void* d_out, int out_size, void* d_ws, size_t ws_size,
                              hipStream_t stream) {
    (void)in_sizes; (void)n_in; (void)d_ws; (void)ws_size; (void)out_size;
    const float* x  = (const float*)d_in[0];
    const float* w0 = (const float*)d_in[1];   // w_self
    const float* w1 = (const float*)d_in[2];   // w_tp
    const float* w2 = (const float*)d_in[3];   // w_tm
    const float* w3 = (const float*)d_in[4];   // w_rp
    const float* w4 = (const float*)d_in[5];   // w_rm
    const float* w5 = (const float*)d_in[6];   // w_cp
    const float* w6 = (const float*)d_in[7];   // w_cm
    float* out = (float*)d_out;

    dim3 grid(B_ * S_ / 128);  // 2048 blocks: 128 positions (4 rows) per block
    dim3 block(128);           // 4 waves of 32
    hipLaunchKernelGGL(dconv3d_wmma_bf16, grid, block, 0, stream,
                       x, w0, w1, w2, w3, w4, w5, w6, out);
}
